// MultiheadedAttention2_40407052321362
// MI455X (gfx1250) — compile-verified, matrix operand order checked by reading, not run on hardware
//
#include <hip/hip_runtime.h>
#include <hip/hip_bf16.h>

// ---------------------------------------------------------------------------
// MultiheadedAttention (batch-axis softmax quirk) for MI455X / gfx1250.
// fp32 -> bf16 once, all 6 GEMMs via v_wmma_f32_16x16x32_bf16 (f32 accum).
// GEMMs are all C = A(MxK) * B(NxK)^T so both operands are K-contiguous.
// Round 2: 128x128 block tile (8 waves), 32x64 wave tile (2x4 WMMA frags),
// register double-buffered K pipeline to overlap L2 loads with WMMA.
// ---------------------------------------------------------------------------

typedef __attribute__((ext_vector_type(16))) __bf16 v16bf;
typedef __attribute__((ext_vector_type(8)))  __bf16 v8bf;
typedef __attribute__((ext_vector_type(8)))  float  v8f;

union Frag16 { v16bf v; v8bf h[2]; };
struct Frags { Frag16 a[2]; Frag16 b[4]; };

#define B_  8
#define T_  2048
#define E_  512
#define HE_ 4096
#define BT_ (B_ * T_)
#define SCALE_ 0.04419417382415922f  // 512^-0.5 (reference scales by E, not HE)

// ------------------------------ fp32 -> bf16 -------------------------------
__global__ void cvt_f32_bf16(const float* __restrict__ in,
                             __bf16* __restrict__ out, size_t n) {
  size_t i = (size_t)blockIdx.x * blockDim.x + threadIdx.x;
  size_t stride = (size_t)gridDim.x * blockDim.x;
  for (; i < n; i += stride) out[i] = (__bf16)in[i];
}

// --------------------------- generic WMMA GEMM -----------------------------
// C[z] = A[z](MxK) * B[z](NxK)^T * scale (+ bias)
// Block = 8 waves (256 thr), 128x128 tile. Wave (mw, nw) in a 4x2 grid owns a
// 32x64 region: 2 A-fragments x 4 B-fragments -> 8 accumulators, with
// register double-buffering across 32-wide K steps.
// OUTMODE: 0 = f32 row-major, 1 = bf16 row-major, 2 = bf16 transposed (C[n][m])
template <int OUTMODE, bool HAS_BIAS>
__global__ __launch_bounds__(256) void gemm_bt(
    const __bf16* __restrict__ A, const __bf16* __restrict__ Bm,
    void* __restrict__ Cp, const float* __restrict__ bias,
    int lda, int ldb, int ldc, int K, float scale,
    long long sA, long long sB, long long sC) {
  const int lane  = threadIdx.x & 31;
  const int wave  = threadIdx.x >> 5;   // 0..7
  const int mw    = wave & 3;           // 4 waves along M
  const int nw    = wave >> 2;          // 2 waves along N
  const int row   = lane & 15;          // A: M row / B: N column / C: N column
  const int khalf = lane >> 4;          // which K half this lane holds

  const size_t z = blockIdx.z;
  A  += z * (size_t)sA;
  Bm += z * (size_t)sB;

  const int m0 = blockIdx.x * 128 + mw * 32;
  const int n0 = blockIdx.y * 128 + nw * 64;

  // Lane base pointers: two 8-elem (16B) chunks per 32-wide K step,
  // at k + khalf*8 and k + khalf*8 + 16 (ISA 16-bit A-matrix layout).
  const __bf16* ap  = A  + (size_t)(m0 + row) * lda + khalf * 8;
  const __bf16* bp0 = Bm + (size_t)(n0 + row) * ldb + khalf * 8;

  v8f acc[2][4] = {};

  auto load_frags = [&](int k, Frags& f) {
#pragma unroll
    for (int mt = 0; mt < 2; ++mt) {
      const __bf16* p = ap + (size_t)(mt * 16) * lda + k;
      f.a[mt].h[0] = *(const v8bf*)(p);
      f.a[mt].h[1] = *(const v8bf*)(p + 16);
    }
#pragma unroll
    for (int nt = 0; nt < 4; ++nt) {
      const __bf16* p = bp0 + (size_t)(nt * 16) * ldb + k;
      f.b[nt].h[0] = *(const v8bf*)(p);
      f.b[nt].h[1] = *(const v8bf*)(p + 16);
    }
  };
  auto compute = [&](const Frags& f) {
#pragma unroll
    for (int mt = 0; mt < 2; ++mt)
#pragma unroll
      for (int nt = 0; nt < 4; ++nt)
        acc[mt][nt] = __builtin_amdgcn_wmma_f32_16x16x32_bf16(
            false, f.a[mt].v, false, f.b[nt].v,
            (short)0, acc[mt][nt], false, false);
  };

  // Software-pipelined K loop: loads for step k+32 overlap WMMAs for step k.
  Frags cur, nxt;
  load_frags(0, cur);
  int k = 0;
  for (; k + 32 < K; k += 32) {
    load_frags(k + 32, nxt);
    compute(cur);
    cur = nxt;
  }
  compute(cur);

  // C fragment: VGPR r -> (M = base + r + 8*khalf, N = base + (lane&15))
#pragma unroll
  for (int nt = 0; nt < 4; ++nt) {
    const int ng = n0 + nt * 16 + row;
    const float bv = HAS_BIAS ? bias[ng] : 0.0f;
#pragma unroll
    for (int mt = 0; mt < 2; ++mt) {
#pragma unroll
      for (int r = 0; r < 8; ++r) {
        const int mg = m0 + mt * 16 + r + 8 * khalf;
        const float val = acc[mt][nt][r] * scale + bv;
        if (OUTMODE == 0) {
          ((float*)Cp)[z * (size_t)sC + (size_t)mg * ldc + ng] = val;
        } else if (OUTMODE == 1) {
          ((__bf16*)Cp)[z * (size_t)sC + (size_t)mg * ldc + ng] = (__bf16)val;
        } else {  // transposed bf16: C[n][m]
          ((__bf16*)Cp)[z * (size_t)sC + (size_t)ng * ldc + mg] = (__bf16)val;
        }
      }
    }
  }
}

// ---------------------- softmax across the BATCH axis ----------------------
__global__ void softmax_batch(const float* __restrict__ scores,
                              __bf16* __restrict__ att, size_t TT) {
  size_t i = (size_t)blockIdx.x * blockDim.x + threadIdx.x;
  size_t stride = (size_t)gridDim.x * blockDim.x;
  for (; i < TT; i += stride) {
    float v[B_];
    float mx = -3.402823466e+38f;
#pragma unroll
    for (int b = 0; b < B_; ++b) {
      v[b] = scores[(size_t)b * TT + i];
      mx = fmaxf(mx, v[b]);
    }
    float s = 0.0f;
#pragma unroll
    for (int b = 0; b < B_; ++b) {
      v[b] = __expf(v[b] - mx);
      s += v[b];
    }
    const float inv = 1.0f / s;
#pragma unroll
    for (int b = 0; b < B_; ++b)
      att[(size_t)b * TT + i] = (__bf16)(v[b] * inv);
  }
}

// ---------------------------------------------------------------------------
extern "C" void kernel_launch(void* const* d_in, const int* in_sizes, int n_in,
                              void* d_out, int out_size, void* d_ws,
                              size_t ws_size, hipStream_t stream) {
  (void)in_sizes; (void)n_in; (void)out_size; (void)ws_size;
  const float* key   = (const float*)d_in[0];
  const float* value = (const float*)d_in[1];
  const float* query = (const float*)d_in[2];
  const float* Wk = (const float*)d_in[3];
  const float* bk = (const float*)d_in[4];
  const float* Wv = (const float*)d_in[5];
  const float* bv = (const float*)d_in[6];
  const float* Wq = (const float*)d_in[7];
  const float* bq = (const float*)d_in[8];
  const float* Wo = (const float*)d_in[9];
  const float* bo = (const float*)d_in[10];

  char* ws = (char*)d_ws;
  const size_t MB = 1024ull * 1024ull;
  __bf16* keyB = (__bf16*)(ws + 0 * MB);     // 16 MB  [B*T, E]
  __bf16* valB = (__bf16*)(ws + 16 * MB);    // 16 MB
  __bf16* qryB = (__bf16*)(ws + 32 * MB);    // 16 MB
  __bf16* WkB  = (__bf16*)(ws + 48 * MB);    //  4 MB  [HE, E]
  __bf16* WvB  = (__bf16*)(ws + 52 * MB);    //  4 MB
  __bf16* WqB  = (__bf16*)(ws + 56 * MB);    //  4 MB
  __bf16* WoB  = (__bf16*)(ws + 60 * MB);    //  4 MB  [E, HE]
  __bf16* qp   = (__bf16*)(ws + 64 * MB);    // 128 MB [B*T, HE]
  __bf16* kp   = (__bf16*)(ws + 192 * MB);   // 128 MB [B*T, HE]
  __bf16* vT   = (__bf16*)(ws + 320 * MB);   // 128 MB [B][HE][T] (transposed)
  float*  sc   = (float*)(ws + 448 * MB);    // 128 MB [B][T][T] scores f32
  __bf16* att  = qp;                         // reuse (qp dead after scores)
  __bf16* attn = kp;                         // reuse (kp dead after softmax)

  const size_t nAct = (size_t)BT_ * E_;
  const size_t nW   = (size_t)HE_ * E_;
  cvt_f32_bf16<<<4096, 256, 0, stream>>>(key,   keyB, nAct);
  cvt_f32_bf16<<<4096, 256, 0, stream>>>(value, valB, nAct);
  cvt_f32_bf16<<<4096, 256, 0, stream>>>(query, qryB, nAct);
  cvt_f32_bf16<<<2048, 256, 0, stream>>>(Wk, WkB, nW);
  cvt_f32_bf16<<<2048, 256, 0, stream>>>(Wv, WvB, nW);
  cvt_f32_bf16<<<2048, 256, 0, stream>>>(Wq, WqB, nW);
  cvt_f32_bf16<<<2048, 256, 0, stream>>>(Wo, WoB, nW);

  const dim3 blk(256);
  // q / k projections: [16384, 512] x [4096, 512]^T -> bf16 [16384, 4096]
  gemm_bt<1, true><<<dim3(BT_ / 128, HE_ / 128, 1), blk, 0, stream>>>(
      qryB, WqB, qp, bq, E_, E_, HE_, E_, 1.0f, 0, 0, 0);
  gemm_bt<1, true><<<dim3(BT_ / 128, HE_ / 128, 1), blk, 0, stream>>>(
      keyB, WkB, kp, bk, E_, E_, HE_, E_, 1.0f, 0, 0, 0);
  // v projection, stored transposed per batch: vT[b][he][t]
  gemm_bt<2, true><<<dim3(T_ / 128, HE_ / 128, B_), blk, 0, stream>>>(
      valB, WvB, vT, bv, E_, E_, T_, E_, 1.0f,
      (long long)T_ * E_, 0, (long long)HE_ * T_);
  // scores[b] = q[b] * k[b]^T * SCALE  (f32 out; per-batch slices L2-resident)
  gemm_bt<0, false><<<dim3(T_ / 128, T_ / 128, B_), blk, 0, stream>>>(
      qp, kp, sc, nullptr, HE_, HE_, T_, HE_, SCALE_,
      (long long)T_ * HE_, (long long)T_ * HE_, (long long)T_ * T_);
  // softmax across batch axis (reference quirk: axis=0)
  softmax_batch<<<8192, 256, 0, stream>>>(sc, att, (size_t)T_ * T_);
  // attended[b] = att[b](TxT) * vT[b](HExT)^T -> bf16 [B*T, HE]
  gemm_bt<1, false><<<dim3(T_ / 128, HE_ / 128, B_), blk, 0, stream>>>(
      att, vT, attn, nullptr, T_, T_, HE_, T_, 1.0f,
      (long long)T_ * T_, (long long)HE_ * T_, (long long)T_ * HE_);
  // out = attended * Wo^T + bo -> f32 [B*T, E]
  gemm_bt<0, true><<<dim3(BT_ / 128, E_ / 128, 1), blk, 0, stream>>>(
      attn, WoB, d_out, bo, HE_, HE_, E_, HE_, 1.0f, 0, 0, 0);
}
